// TransferNet_30640296689802
// MI455X (gfx1250) — matrix operand order repaired and checked by reading, not compiled, not run on hardware
//
#include <hip/hip_runtime.h>
#include <hip/hip_bf16.h>
#include <math.h>
#include <stdint.h>

#define NEnt  100000
#define NRel  400
#define NTrip 1000000
#define BB    32
#define DHH   512
#define TT    3
#define SELFREL 0

typedef __attribute__((ext_vector_type(16))) __bf16 v16bf;
typedef __attribute__((ext_vector_type(8)))  float  v8f;

__device__ __forceinline__ v8f wmma_bf16(v16bf a, v16bf b, v8f c) {
  // v_wmma_f32_16x16x32_bf16  (neg_a, A, neg_b, B, c_mod, C, reuse_a, reuse_b)
  return __builtin_amdgcn_wmma_f32_16x16x32_bf16(false, a, false, b, (short)0, c, false, false);
}

// ---- async global->LDS staging (gfx1250 ASYNCcnt path), with sync fallback ----
#if defined(__AMDGCN__) && __has_builtin(__builtin_amdgcn_global_load_async_to_lds_b128)
#define ASYNC_LDS_COPY 1
#define AS1 __attribute__((address_space(1)))
#define AS3 __attribute__((address_space(3)))
typedef int v4i_vs __attribute__((vector_size(16)));   // matches builtin's pointee type
#if __has_builtin(__builtin_amdgcn_s_wait_asynccnt)
#define WAIT_ASYNC_8() __builtin_amdgcn_s_wait_asynccnt(8)
#define WAIT_ASYNC_0() __builtin_amdgcn_s_wait_asynccnt(0)
#else
#define WAIT_ASYNC_8() asm volatile("s_wait_asynccnt 0x8" ::: "memory")
#define WAIT_ASYNC_0() asm volatile("s_wait_asynccnt 0x0" ::: "memory")
#endif
#else
#define WAIT_ASYNC_8()
#define WAIT_ASYNC_0()
#endif

// copy one contiguous 32x512 f32 slab (64KB) to LDS; 512 threads x 8 x 16B chunks
__device__ __forceinline__ void stage_tile(const float* __restrict__ gsrc, float* lds_dst, int tid) {
#ifdef ASYNC_LDS_COPY
#pragma unroll
  for (int t = 0; t < 8; ++t) {
    int c = tid + t * 512;                 // 0..4095 16-byte chunks
    __builtin_amdgcn_global_load_async_to_lds_b128(
        (AS1 v4i_vs*)(uintptr_t)(gsrc + c * 4),
        (AS3 v4i_vs*)(uint32_t)(uintptr_t)(lds_dst + c * 4),
        0, 0);
  }
#else
#pragma unroll
  for (int t = 0; t < 8; ++t) {
    int c = tid + t * 512;
    *(float4*)(lds_dst + c * 4) = *(const float4*)(gsrc + c * 4);
  }
#endif
}

// ---------- small utility kernels ----------
__global__ void zero_kernel(float* p, int count) {
  int i = blockIdx.x * 256 + threadIdx.x;
  if (i < count) p[i] = 0.f;
}

// last_e[n*32+b] = e_s[b*NE+n]   (entity-major internal layout)
__global__ void init_laste(const float* __restrict__ e_s, float* __restrict__ last_e) {
  int idx = blockIdx.x * 256 + threadIdx.x;       // < NE*B
  int n = idx >> 5, b = idx & 31;
  last_e[idx] = e_s[(size_t)b * NEnt + n];
}

__global__ void qemb_kernel(const int* __restrict__ questions,
                            const float* __restrict__ rel_emb,
                            float* __restrict__ qemb) {
  int idx = blockIdx.x * 256 + threadIdx.x;       // < B*DH
  int b = idx >> 9, d = idx & 511;
  qemb[idx] = rel_emb[(size_t)questions[b] * DHH + d];
}

// cq[b][j] = tanh( sum_k qemb[b][k] * W[j][k] + bvec[j] ),  blockDim=256, grid=DH/8
__global__ void proj_kernel(const float* __restrict__ qemb, const float* __restrict__ W,
                            const float* __restrict__ bvec, float* __restrict__ cq) {
  __shared__ float qs[32 * 128];
  int jl = threadIdx.x >> 5;
  int b  = threadIdx.x & 31;
  int j  = blockIdx.x * 8 + jl;
  float acc = 0.f;
  for (int kc = 0; kc < 4; ++kc) {
    __syncthreads();
    for (int i = threadIdx.x; i < 32 * 128; i += 256)
      qs[i] = qemb[(size_t)(i >> 7) * DHH + kc * 128 + (i & 127)];
    __syncthreads();
    const float* wrow = W + (size_t)j * DHH + kc * 128;
    const float* qrow = qs + b * 128;
#pragma unroll 8
    for (int kk = 0; kk < 128; ++kk) acc = fmaf(qrow[kk], wrow[kk], acc);
  }
  cq[(size_t)b * DHH + j] = tanhf(acc + bvec[j]);
}

// dist[b][r] = dot(cq[b], rel_emb[r]) ; grid = B*NR/256
__global__ void logits_kernel(const float* __restrict__ cq, const float* __restrict__ rel_emb,
                              float* __restrict__ dist) {
  int tid = blockIdx.x * 256 + threadIdx.x;
  int b = tid & 31, r = tid >> 5;
  if (r >= NRel) return;
  const float* crow = cq + (size_t)b * DHH;
  const float* rrow = rel_emb + (size_t)r * DHH;
  float acc = 0.f;
#pragma unroll 8
  for (int k = 0; k < DHH; ++k) acc = fmaf(crow[k], rrow[k], acc);
  dist[(size_t)b * NRel + r] = acc;
}

// in-place softmax per row + argmax + (t==0) gt extraction ; grid = B, blockDim = 128
__global__ void softmax_kernel(float* dist, int* amax, float* gt,
                               const int* __restrict__ questions, int t) {
  int b = blockIdx.x;
  __shared__ float sval[NRel];
  __shared__ float sbc[2];
  for (int r = threadIdx.x; r < NRel; r += 128) sval[r] = dist[(size_t)b * NRel + r];
  __syncthreads();
  if (threadIdx.x == 0) {
    float mx = sval[0]; int mi = 0;
    for (int r = 1; r < NRel; ++r) if (sval[r] > mx) { mx = sval[r]; mi = r; }
    float s = 0.f;
    for (int r = 0; r < NRel; ++r) s += expf(sval[r] - mx);
    sbc[0] = mx; sbc[1] = s;
    amax[t * BB + b] = mi;
    if (t == 0) gt[b] = expf(sval[questions[b]] - mx) / s;
  }
  __syncthreads();
  float mx = sbc[0], s = sbc[1];
  for (int r = threadIdx.x; r < NRel; r += 128)
    dist[(size_t)b * NRel + r] = expf(sval[r] - mx) / s;
}

// new_e[obj*32+b] += last_e[subj*32+b] * dist[b][rel]  ; NT*32 threads
__global__ void follow_kernel(const float* __restrict__ last_e, const float* __restrict__ dist,
                              const int* __restrict__ subj, const int* __restrict__ rel,
                              const int* __restrict__ obj, float* new_e) {
  long tid = (long)blockIdx.x * 256 + threadIdx.x;
  int i = (int)(tid >> 5);
  int b = (int)(tid & 31);
  if (i >= NTrip) return;
  float x = last_e[(size_t)subj[i] * BB + b] * dist[(size_t)b * NRel + rel[i]];
  atomicAdd(&new_e[(size_t)obj[i] * BB + b], x);
}

// fused per-step elementwise postprocessing; grid = NE*B/256
__global__ void post_kernel(int t, const float* __restrict__ new_e, float* __restrict__ last_e,
                            float* __restrict__ snap1, const float* __restrict__ answers,
                            const float* __restrict__ gt, const float* __restrict__ e_s,
                            const int* __restrict__ amax, const float* __restrict__ prev2_em,
                            const float* __restrict__ prev2_bm) {
  int idx = blockIdx.x * 256 + threadIdx.x;
  int n = idx >> 5, b = idx & 31;
  float v = new_e[idx];
  if (t == 0) v -= answers[(size_t)b * NEnt + n] * gt[b];
  if (v > 1.0f) v = 1.0f;                 // z-clamp: x/z with z = x (x>1) else 1
  if (t > 0) {
    int a1 = amax[(t - 1) * BB + b], a2 = amax[t * BB + b];
    int d = a1 - a2; if (d < 0) d = -d;
    int mn = a1 < a2 ? a1 : a2;
    if (d == 1 && (mn & 1) == 0) {
      float p2 = prev2_em ? prev2_em[idx] : prev2_bm[(size_t)b * NEnt + n];
      if (p2 > 0.9f) v = 0.f;
    }
  }
  if (t == TT - 1) {
    int ns = (amax[0 * BB + b] == SELFREL) + (amax[1 * BB + b] == SELFREL) +
             (amax[2 * BB + b] == SELFREL);
    if (ns == 1) v *= (1.0f - e_s[(size_t)b * NEnt + n]);
  }
  last_e[idx] = v;
  if (t == 0) snap1[idx] = v;
}

// write e_score (B,NE) + per-row sums ; grid = 1024, blockDim = 256
__global__ void finalize_kernel(const float* __restrict__ e, float* __restrict__ out_escore,
                                float* rowsum) {
  __shared__ float red[256];
  int b = threadIdx.x & 31;
  float local = 0.f;
  long total = (long)NEnt * BB;
  for (long idx = (long)blockIdx.x * 256 + threadIdx.x; idx < total; idx += (long)gridDim.x * 256) {
    int n = (int)(idx >> 5);
    float v = e[idx];
    out_escore[(size_t)b * NEnt + n] = v;   // idx&31 == threadIdx.x&31 (stride mult of 32)
    local += v;
  }
  red[threadIdx.x] = local;
  __syncthreads();
  if (threadIdx.x < 32) {
    float s = 0.f;
    for (int w = 0; w < 8; ++w) s += red[w * 32 + threadIdx.x];
    atomicAdd(&rowsum[threadIdx.x], s);
  }
}

// GEMM1: svec[b][d] += sum_n normed[n,b] * E[n][d]
// M=32, N=512 over 16 waves, split-K grid of 25; B-slabs double-buffered through LDS
// via async global->LDS (ASYNCcnt) when available. Dynamic LDS: 2 x 64KB.
__global__ __launch_bounds__(512) void gemm1_kernel(const float* __restrict__ e,
                                                    const float* __restrict__ rowsum,
                                                    const float* __restrict__ ent_emb,
                                                    float* svec) {
  extern __shared__ float sB[];              // 2 * 16384 floats
  int tid = threadIdx.x;
  int lane = tid & 31;
  int w = tid >> 5;                          // 0..15 -> 32 columns each (2 N-tiles)
  int colBase = w * 32;
  int k0 = blockIdx.x * 4000;                // 25 * 4000 = NE, 4000 % 32 == 0
  int kend = k0 + 4000;
  int mrow = lane & 15;
  int nlo = lane & 15;
  int rbase = (lane >> 4) << 4;              // 0 or 16 (unused; B tile uses table k order)
  float inv0 = 1.0f / (rowsum[mrow] + 1e-6f);
  float inv1 = 1.0f / (rowsum[16 + mrow] + 1e-6f);
  v8f a00 = {}, a01 = {}, a10 = {}, a11 = {};

  stage_tile(ent_emb + (size_t)k0 * DHH, sB, tid);   // prologue into buffer 0
  int p = 0;
  for (int k = k0; k < kend; k += 32) {
    float* cur = sB + p * 16384;
    float* nxt = sB + (p ^ 1) * 16384;
    if (k + 32 < kend) { stage_tile(ent_emb + (size_t)(k + 32) * DHH, nxt, tid); WAIT_ASYNC_8(); }
    else               { WAIT_ASYNC_0(); }
    __syncthreads();

    v16bf A0, A1, B0, B1;
#pragma unroll
    for (int h = 0; h < 16; ++h) {
      int K = ((h >> 3) << 4) + ((lane >> 4) << 3) + (h & 7);   // A 16x32 bf16 layout
      A0[h] = (__bf16)(e[(size_t)(k + K) * BB + mrow] * inv0);
      A1[h] = (__bf16)(e[(size_t)(k + K) * BB + 16 + mrow] * inv1);
    }
#pragma unroll
    for (int h = 0; h < 16; ++h) {                              // B 32x16 bf16 layout (from LDS)
      const float* rowp = cur + (((h >> 3) << 4) + ((lane >> 4) << 3) + (h & 7)) * DHH;
      B0[h] = (__bf16)rowp[colBase + nlo];
      B1[h] = (__bf16)rowp[colBase + 16 + nlo];
    }
    a00 = wmma_bf16(A0, B0, a00);
    a01 = wmma_bf16(A0, B1, a01);
    a10 = wmma_bf16(A1, B0, a10);
    a11 = wmma_bf16(A1, B1, a11);

    __syncthreads();                         // all reads of `cur` done before it is restaged
    p ^= 1;
  }

  int mAdd = (lane < 16) ? 0 : 8;
#pragma unroll
  for (int r = 0; r < 8; ++r) {
    int M0 = r + mAdd;
    atomicAdd(&svec[(size_t)M0 * DHH + colBase + nlo],             a00[r]);
    atomicAdd(&svec[(size_t)M0 * DHH + colBase + 16 + nlo],        a01[r]);
    atomicAdd(&svec[(size_t)(16 + M0) * DHH + colBase + nlo],      a10[r]);
    atomicAdd(&svec[(size_t)(16 + M0) * DHH + colBase + 16 + nlo], a11[r]);
  }
}

// GEMM2: pred[b][n] = sum_d svec[b][d]*E[n][d] + bias[n]  (one 16-entity tile per wave)
__global__ __launch_bounds__(256) void gemm2_kernel(const float* __restrict__ svec,
                                                    const float* __restrict__ ent_emb,
                                                    const float* __restrict__ ent_bias,
                                                    float* __restrict__ out) {
  __shared__ __bf16 sA[32 * DHH];                 // 32KB in the 320KB WGP LDS
  for (int i = threadIdx.x; i < 32 * DHH; i += 256) sA[i] = (__bf16)svec[i];
  __syncthreads();
  int lane = threadIdx.x & 31;
  int w = threadIdx.x >> 5;                        // 0..7
  int g = blockIdx.x * 8 + w;
  if (g >= NEnt / 16) return;                      // no barriers after this point
  int n0 = g * 16;
  int nl = lane & 15;
  v8f acc0 = {}, acc1 = {};
  for (int k0 = 0; k0 < DHH; k0 += 32) {
    v16bf A0, A1, Bt;
#pragma unroll
    for (int h = 0; h < 16; ++h) {
      int K = ((lane >> 4) << 4) + h;
      A0[h] = sA[nl * DHH + k0 + K];
      A1[h] = sA[(16 + nl) * DHH + k0 + K];
    }
    const float4* bp = (const float4*)(ent_emb + (size_t)(n0 + nl) * DHH + k0 + ((lane >> 4) << 4));
    float4 f0 = bp[0], f1 = bp[1], f2 = bp[2], f3 = bp[3];
    Bt[0] = (__bf16)f0.x;  Bt[1] = (__bf16)f0.y;  Bt[2] = (__bf16)f0.z;  Bt[3] = (__bf16)f0.w;
    Bt[4] = (__bf16)f1.x;  Bt[5] = (__bf16)f1.y;  Bt[6] = (__bf16)f1.z;  Bt[7] = (__bf16)f1.w;
    Bt[8] = (__bf16)f2.x;  Bt[9] = (__bf16)f2.y;  Bt[10] = (__bf16)f2.z; Bt[11] = (__bf16)f2.w;
    Bt[12] = (__bf16)f3.x; Bt[13] = (__bf16)f3.y; Bt[14] = (__bf16)f3.z; Bt[15] = (__bf16)f3.w;
    acc0 = wmma_bf16(A0, Bt, acc0);
    acc1 = wmma_bf16(A1, Bt, acc1);
  }
  float bias = ent_bias[n0 + nl];
  int mAdd = (lane < 16) ? 0 : 8;
#pragma unroll
  for (int r = 0; r < 8; ++r) {
    out[(size_t)(r + mAdd) * NEnt + n0 + nl]      = acc0[r] + bias;
    out[(size_t)(16 + r + mAdd) * NEnt + n0 + nl] = acc1[r] + bias;
  }
}

extern "C" void kernel_launch(void* const* d_in, const int* in_sizes, int n_in,
                              void* d_out, int out_size, void* d_ws, size_t ws_size,
                              hipStream_t stream) {
  (void)in_sizes; (void)n_in; (void)out_size;
  const int*   questions = (const int*)d_in[0];
  const float* e_s       = (const float*)d_in[1];
  const float* answers   = (const float*)d_in[2];
  const int*   subj      = (const int*)d_in[3];
  const int*   rel       = (const int*)d_in[4];
  const int*   obj       = (const int*)d_in[5];
  const float* rel_emb   = (const float*)d_in[6];
  const float* step_W    = (const float*)d_in[7];
  const float* step_b    = (const float*)d_in[8];
  const float* ent_emb   = (const float*)d_in[9];
  const float* ent_bias  = (const float*)d_in[10];
  float* out = (float*)d_out;

  const size_t NEB = (size_t)NEnt * BB;            // 3.2M floats
  size_t need = (3 * NEB + 2 * BB * DHH + BB * NRel + 2 * BB + BB * DHH + TT * BB + 64) * 4;
  if (ws_size < need) return;

  float* ws     = (float*)d_ws;
  float* last_e = ws;
  float* new_e  = ws + NEB;
  float* snap1  = ws + 2 * NEB;
  float* qemb   = ws + 3 * NEB;
  float* cq     = qemb + BB * DHH;
  float* dist   = cq + BB * DHH;
  float* gt     = dist + BB * NRel;
  float* rowsum = gt + BB;
  float* svec   = rowsum + BB;                     // contiguous with rowsum
  int*   amax   = (int*)(svec + BB * DHH);

  init_laste<<<(int)(NEB / 256), 256, 0, stream>>>(e_s, last_e);
  zero_kernel<<<(BB + BB * DHH + 255) / 256, 256, 0, stream>>>(rowsum, BB + BB * DHH);
  qemb_kernel<<<(BB * DHH) / 256, 256, 0, stream>>>(questions, rel_emb, qemb);

  for (int t = 0; t < TT; ++t) {
    proj_kernel<<<DHH / 8, 256, 0, stream>>>(qemb, step_W + (size_t)t * DHH * DHH,
                                             step_b + (size_t)t * DHH, cq);
    logits_kernel<<<(BB * NRel) / 256, 256, 0, stream>>>(cq, rel_emb, dist);
    softmax_kernel<<<BB, 128, 0, stream>>>(dist, amax, gt, questions, t);
    zero_kernel<<<(int)(NEB / 256), 256, 0, stream>>>(new_e, (int)NEB);
    follow_kernel<<<(NTrip * 32) / 256, 256, 0, stream>>>(last_e, dist, subj, rel, obj, new_e);
    const float* p2em = (t == 2) ? snap1 : nullptr;
    const float* p2bm = (t == 1) ? e_s : nullptr;
    post_kernel<<<(int)(NEB / 256), 256, 0, stream>>>(t, new_e, last_e, snap1, answers, gt,
                                                      e_s, amax, p2em, p2bm);
  }

  finalize_kernel<<<1024, 256, 0, stream>>>(last_e, out, rowsum);
  gemm1_kernel<<<25, 512, 2 * 16384 * sizeof(float), stream>>>(last_e, rowsum, ent_emb, svec);
  gemm2_kernel<<<(NEnt / 16 + 7) / 8, 256, 0, stream>>>(svec, ent_emb, ent_bias, out + NEB);
}
